// RetinaNet_42537356099978
// MI455X (gfx1250) — matrix operand order repaired and checked by reading, not compiled, not run on hardware
//
#include <hip/hip_runtime.h>
#include <hip/hip_bf16.h>
#include <math.h>

typedef __attribute__((ext_vector_type(16))) _Float16 v16h;
typedef __attribute__((ext_vector_type(8)))  _Float16 v8h;
typedef __attribute__((ext_vector_type(8)))  float    v8f;

#define CIN   256
#define KTOT  2304      // 256 * 9
#define MAXC  4096      // candidate cap per image
#define NIMG  2
#define NLEV  5
#define TPX   64        // spatial pixels per block (4 WMMA N-fragments)

// ---------------------------------------------------------------- utilities
__global__ void f32_to_f16_kernel(const float* __restrict__ src,
                                  _Float16* __restrict__ dst, int n) {
  int i = blockIdx.x * 256 + threadIdx.x;
  if (i < n) dst[i] = (_Float16)src[i];
}

__global__ void zero_counts_kernel(int* c) {
  if (threadIdx.x < NIMG) c[threadIdx.x] = 0;
}

// ---------------------------------------------------------------- conv3x3 via WMMA implicit GEMM
// Block: 256 threads = 8 waves, each wave owns one 16-channel co tile.
// Tile: [16 co x 64 px], K = CIN*9 = 2304, stepped 32 at a time.
// LDS: double-buffered 64x32 f16 im2col tile (row stride 48 halves for
// 32B-aligned v16h loads and bank spread).
// MODE 0: bias + ReLU -> f16 activation buffer
// MODE 1: bias + sigmoid + threshold -> candidate append (score head, COUT=720)
// MODE 2: bias -> f32 reg buffer (reg head, COUT=36)
template <int MODE, int COUT, typename TIN>
__global__ __launch_bounds__(256)
void conv3x3_wmma_kernel(const TIN* __restrict__ in, const _Float16* __restrict__ wgt,
                         const float* __restrict__ bias, void* __restrict__ outp,
                         int H, int W, int logW,
                         int level, int regOffF,
                         int* __restrict__ cand_count, int* __restrict__ cand_info,
                         float* __restrict__ cand_score) {
  const int HW      = H * W;
  const int tid     = threadIdx.x;
  const int img     = blockIdx.z;
  const int sp_base = blockIdx.x * TPX;
  const int wave    = tid >> 5;
  const int lane    = tid & 31;
  const int co_base = (blockIdx.y * 8 + wave) * 16;
  const bool active = (co_base < COUT);

  __shared__ alignas(32) _Float16 ldsB[2][TPX * 48];

  const TIN* inN = in + (size_t)img * CIN * HW;

  // ---- im2col fill: each thread owns one fixed k (tid&31) and 8 pixels
  const int fk  = tid & 31;
  const int fn0 = tid >> 5;
  auto fill = [&](int buf, int kb) {
    int kk = kb + fk;
    int ci = kk / 9;
    int rr = kk - ci * 9;
    int ky = rr / 3;
    int kx = rr - ky * 3;
    const TIN* base = inN + (size_t)ci * HW;
#pragma unroll
    for (int i = 0; i < 8; ++i) {
      int n  = fn0 + 8 * i;
      int p  = sp_base + n;
      int y  = p >> logW;
      int x  = p & (W - 1);
      int iy = y + ky - 1;
      int ix = x + kx - 1;
      _Float16 v = (_Float16)0.0f;
      if (iy >= 0 && iy < H && ix >= 0 && ix < W)
        v = (_Float16)(float)base[iy * W + ix];
      ldsB[buf][n * 48 + fk] = v;
    }
  };

  v8f acc[4] = {};

  // fragment addressing constants
  const int nn   = lane & 15;
  const int koff = (lane < 16) ? 0 : 16;
  const int ksh  = (lane < 16) ? 0 : 8;
  int co_a = co_base + (lane & 15);
  if (COUT % 16 && co_a >= COUT) co_a = COUT - 1;  // folds away for 256/720
  const _Float16* wrow0 = wgt + (size_t)co_a * KTOT;

  fill(0, 0);
  __syncthreads();

  for (int kb = 0; kb < KTOT; kb += 32) {
    const int cur = (kb >> 5) & 1;
    if (kb + 32 < KTOT) fill(cur ^ 1, kb + 32);  // prefetch next K-slice

    if (active) {
      // A fragment: lanes 0-15 M=lane K{0..15}; lanes 16-31 M=lane-16 K{16..31}
      const _Float16* wrow = wrow0 + kb;
      v16h a = *(const v16h*)(wrow + koff);

      // issue all B-fragment LDS loads first, then one wait + 4 back-to-back WMMAs
      v16h b[4];
#pragma unroll
      for (int f = 0; f < 4; ++f)
        b[f] = *(const v16h*)&ldsB[cur][(f * 16 + nn) * 48 + koff];
#pragma unroll
      for (int f = 0; f < 4; ++f)
        acc[f] = __builtin_amdgcn_wmma_f32_16x16x32_f16(false, a, false, b[f],
                                                        (short)0, acc[f], false, false);
    }
    __syncthreads();
  }

  if (!active) return;

  // C layout: VGPR r -> M = r + 8*(lane>=16), N = lane%16
  const int mb = (lane >> 4) << 3;
#pragma unroll
  for (int f = 0; f < 4; ++f) {
    const int p = sp_base + f * 16 + nn;
#pragma unroll
    for (int r = 0; r < 8; ++r) {
      int co = co_base + mb + r;
      if (COUT % 16 && co >= COUT) continue;  // only reg head (36) keeps this
      float v = acc[f][r] + bias[co];
      if (MODE == 0) {
        _Float16* out = (_Float16*)outp;
        out[(size_t)img * COUT * HW + (size_t)co * HW + p] = (_Float16)fmaxf(v, 0.0f);
      } else if (MODE == 2) {
        float* out = (float*)outp;
        out[(size_t)regOffF + ((size_t)img * 36 + co) * HW + p] = v;
      } else {  // MODE == 1: score head
        float s = 1.0f / (1.0f + expf(-v));
        if (s > 0.05f) {
          int aidx = co / 80;          // channel = a*NUM_CLASSES + cls
          int cls  = co - aidx * 80;
          int pa   = p * 9 + aidx;     // anchor index within level
          int idx  = atomicAdd(&cand_count[img], 1);
          if (idx < MAXC) {
            cand_info[img * MAXC + idx]  = (level << 25) | (cls << 18) | pa;
            cand_score[img * MAXC + idx] = s;
          }
        }
      }
    }
  }
}

// ---------------------------------------------------------------- decode + greedy NMS (one block per image)
__global__ __launch_bounds__(256)
void decode_nms_kernel(const int* __restrict__ cand_count, const int* __restrict__ cand_info,
                       const float* __restrict__ cand_score, const float* __restrict__ regbuf,
                       float* __restrict__ scratch, float* __restrict__ out) {
  const int img = blockIdx.x;
  const int tid = threadIdx.x;
  int n = cand_count[img];
  if (n > MAXC) n = MAXC;

  const int   LW[NLEV]      = {128, 64, 32, 16, 8};
  const int   LLOGW[NLEV]   = {7, 6, 5, 4, 3};
  const float LSTRIDE[NLEV] = {8.f, 16.f, 32.f, 64.f, 128.f};
  const float LSIZE[NLEV]   = {32.f, 64.f, 128.f, 256.f, 512.f};
  const int   ROFF[NLEV]    = {0, 1179648, 1474560, 1548288, 1566720};
  const float SCALE_CLAMP   = 4.135166556742356f;  // log(1000/16)

  float* bx   = scratch + (size_t)img * MAXC * 10;  // decoded boxes
  float* ob   = bx + 4 * MAXC;                      // class-offset boxes
  float* area = ob + 4 * MAXC;
  float* sarr = area + MAXC;                        // live scores

  for (int j = tid; j < n; j += 256) {
    int info = cand_info[img * MAXC + j];
    int lvl = (info >> 25) & 7;
    int cls = (info >> 18) & 127;
    int pa  = info & ((1 << 18) - 1);
    int p   = pa / 9;
    int a   = pa - p * 9;
    int Wl  = LW[lvl], HWl = Wl * Wl;
    int y   = p >> LLOGW[lvl];
    int x   = p & (Wl - 1);
    int si  = a / 3;
    int ari = a - si * 3;
    float scale = LSIZE[lvl] * exp2f((float)si * (1.0f / 3.0f));
    float areaA = scale * scale;
    float ar    = (ari == 0) ? 0.5f : ((ari == 1) ? 1.0f : 2.0f);
    float wa    = sqrtf(areaA / ar);
    float ha    = ar * wa;
    float cxa   = ((float)x + 0.5f) * LSTRIDE[lvl];
    float cya   = ((float)y + 0.5f) * LSTRIDE[lvl];
    const float* rb = regbuf + ROFF[lvl] + (size_t)img * 36 * HWl;
    float dx = rb[(a * 4 + 0) * HWl + p];
    float dy = rb[(a * 4 + 1) * HWl + p];
    float dw = fminf(rb[(a * 4 + 2) * HWl + p], SCALE_CLAMP);
    float dh = fminf(rb[(a * 4 + 3) * HWl + p], SCALE_CLAMP);
    float pcx = dx * wa + cxa;
    float pcy = dy * ha + cya;
    float pw  = expf(dw) * wa;
    float ph  = expf(dh) * ha;
    float x1 = fminf(fmaxf(pcx - 0.5f * pw, 0.f), 1024.f);
    float y1 = fminf(fmaxf(pcy - 0.5f * ph, 0.f), 1024.f);
    float x2 = fminf(fmaxf(pcx + 0.5f * pw, 0.f), 1024.f);
    float y2 = fminf(fmaxf(pcy + 0.5f * ph, 0.f), 1024.f);
    bx[j * 4 + 0] = x1; bx[j * 4 + 1] = y1; bx[j * 4 + 2] = x2; bx[j * 4 + 3] = y2;
    float off = (float)cls * 2048.0f;
    ob[j * 4 + 0] = x1 + off; ob[j * 4 + 1] = y1 + off;
    ob[j * 4 + 2] = x2 + off; ob[j * 4 + 3] = y2 + off;
    area[j] = (x2 - x1) * (y2 - y1);
    sarr[j] = cand_score[img * MAXC + j];
  }
  __syncthreads();

  __shared__ float rs[256];
  __shared__ int   ri[256];
  __shared__ int   rk[256];

  for (int it = 0; it < 100; ++it) {
    // argmax with deterministic tie-break on packed candidate key
    float bs = -1e30f; int bi = -1; int bk = 0x7fffffff;
    for (int j = tid; j < n; j += 256) {
      float sj = sarr[j];
      int   kj = cand_info[img * MAXC + j];
      if (sj > bs || (sj == bs && kj < bk)) { bs = sj; bi = j; bk = kj; }
    }
    rs[tid] = bs; ri[tid] = bi; rk[tid] = bk;
    __syncthreads();
    for (int stp = 128; stp > 0; stp >>= 1) {
      if (tid < stp) {
        float s2 = rs[tid + stp]; int k2 = rk[tid + stp];
        if (s2 > rs[tid] || (s2 == rs[tid] && k2 < rk[tid])) {
          rs[tid] = s2; ri[tid] = ri[tid + stp]; rk[tid] = k2;
        }
      }
      __syncthreads();
    }
    float ms = rs[0]; int mi = ri[0];
    __syncthreads();

    if (ms > 0.0f && mi >= 0) {
      float bx0 = ob[mi * 4 + 0], by0 = ob[mi * 4 + 1];
      float bx1 = ob[mi * 4 + 2], by1 = ob[mi * 4 + 3];
      float ai = area[mi];
      for (int j = tid; j < n; j += 256) {
        float ix1 = fmaxf(bx0, ob[j * 4 + 0]);
        float iy1 = fmaxf(by0, ob[j * 4 + 1]);
        float ix2 = fminf(bx1, ob[j * 4 + 2]);
        float iy2 = fminf(by1, ob[j * 4 + 3]);
        float inter = fmaxf(ix2 - ix1, 0.f) * fmaxf(iy2 - iy1, 0.f);
        float iou = inter / fmaxf(ai + area[j] - inter, 1e-6f);
        if (iou > 0.5f) sarr[j] = -1.0f;   // suppresses mi itself too (IoU=1)
      }
      if (tid == 0) {
        out[img * 400 + it * 4 + 0] = bx[mi * 4 + 0];
        out[img * 400 + it * 4 + 1] = bx[mi * 4 + 1];
        out[img * 400 + it * 4 + 2] = bx[mi * 4 + 2];
        out[img * 400 + it * 4 + 3] = bx[mi * 4 + 3];
        out[800 + img * 100 + it] = ms;
        int info = cand_info[img * MAXC + mi];
        out[1000 + img * 100 + it] = (float)((info >> 18) & 127);
      }
    } else {
      if (tid == 0) {
        out[img * 400 + it * 4 + 0] = 0.f;
        out[img * 400 + it * 4 + 1] = 0.f;
        out[img * 400 + it * 4 + 2] = 0.f;
        out[img * 400 + it * 4 + 3] = 0.f;
        out[800 + img * 100 + it]  = 0.f;
        out[1000 + img * 100 + it] = -1.0f;
      }
    }
    __syncthreads();
  }
}

// ---------------------------------------------------------------- host orchestration
extern "C" void kernel_launch(void* const* d_in, const int* in_sizes, int n_in,
                              void* d_out, int out_size, void* d_ws, size_t ws_size,
                              hipStream_t stream) {
  const float* feat[NLEV];
  for (int i = 0; i < NLEV; ++i) feat[i] = (const float*)d_in[i];
  const float* w_cls   = (const float*)d_in[5];
  const float* b_cls   = (const float*)d_in[6];
  const float* w_box   = (const float*)d_in[7];
  const float* b_box   = (const float*)d_in[8];
  const float* w_score = (const float*)d_in[9];
  const float* b_score = (const float*)d_in[10];
  const float* w_reg   = (const float*)d_in[11];
  const float* b_reg   = (const float*)d_in[12];

  char* ws = (char*)d_ws;
  size_t off = 0;
  auto take = [&](size_t bytes) -> char* {
    char* p = ws + off;
    off = (off + bytes + 255) & ~(size_t)255;
    return p;
  };
  _Float16* wg_cls = (_Float16*)take(4ull * CIN * KTOT * 2);
  _Float16* wg_box = (_Float16*)take(4ull * CIN * KTOT * 2);
  _Float16* wg_sc  = (_Float16*)take(720ull * KTOT * 2);
  _Float16* wg_rg  = (_Float16*)take(36ull * KTOT * 2);
  _Float16* pingA  = (_Float16*)take(2ull * CIN * 16384 * 2);
  _Float16* pingB  = (_Float16*)take(2ull * CIN * 16384 * 2);
  float*    regbuf = (float*)take(1571328ull * 4);
  int*      cc     = (int*)take(NIMG * 4);
  int*      cinfo  = (int*)take((size_t)NIMG * MAXC * 4);
  float*    cscore = (float*)take((size_t)NIMG * MAXC * 4);
  float*    nmss   = (float*)take((size_t)NIMG * MAXC * 10 * 4);

  auto cvt = [&](const float* s, _Float16* d, int n) {
    f32_to_f16_kernel<<<(n + 255) / 256, 256, 0, stream>>>(s, d, n);
  };
  cvt(w_cls, wg_cls, 4 * CIN * KTOT);
  cvt(w_box, wg_box, 4 * CIN * KTOT);
  cvt(w_score, wg_sc, 720 * KTOT);
  cvt(w_reg, wg_rg, 36 * KTOT);
  zero_counts_kernel<<<1, 32, 0, stream>>>(cc);

  const int WL[NLEV]   = {128, 64, 32, 16, 8};
  const int LOGW[NLEV] = {7, 6, 5, 4, 3};
  const int ROFF[NLEV] = {0, 1179648, 1474560, 1548288, 1566720};
  const size_t LAYSTR  = (size_t)CIN * KTOT;  // halves per tower layer

  for (int l = 0; l < NLEV; ++l) {
    const int Wd = WL[l], H = Wd, HW = Wd * Wd, lw = LOGW[l];
    dim3 blk(256);
    dim3 g256(HW / TPX, 2, NIMG);  // 16 co-tiles / 8 waves -> gy=2
    dim3 gsc(HW / TPX, 6, NIMG);   // 45 co-tiles -> gy=6
    dim3 grg(HW / TPX, 1, NIMG);   // 3 co-tiles  -> gy=1

    // classification tower -> score head (candidate append)
    conv3x3_wmma_kernel<0, CIN, float><<<g256, blk, 0, stream>>>(
        feat[l], wg_cls + 0 * LAYSTR, b_cls + 0 * CIN, pingA, H, Wd, lw,
        0, 0, nullptr, nullptr, nullptr);
    conv3x3_wmma_kernel<0, CIN, _Float16><<<g256, blk, 0, stream>>>(
        pingA, wg_cls + 1 * LAYSTR, b_cls + 1 * CIN, pingB, H, Wd, lw,
        0, 0, nullptr, nullptr, nullptr);
    conv3x3_wmma_kernel<0, CIN, _Float16><<<g256, blk, 0, stream>>>(
        pingB, wg_cls + 2 * LAYSTR, b_cls + 2 * CIN, pingA, H, Wd, lw,
        0, 0, nullptr, nullptr, nullptr);
    conv3x3_wmma_kernel<0, CIN, _Float16><<<g256, blk, 0, stream>>>(
        pingA, wg_cls + 3 * LAYSTR, b_cls + 3 * CIN, pingB, H, Wd, lw,
        0, 0, nullptr, nullptr, nullptr);
    conv3x3_wmma_kernel<1, 720, _Float16><<<gsc, blk, 0, stream>>>(
        pingB, wg_sc, b_score, nullptr, H, Wd, lw,
        l, 0, cc, cinfo, cscore);

    // box tower -> reg head (f32 buffer)
    conv3x3_wmma_kernel<0, CIN, float><<<g256, blk, 0, stream>>>(
        feat[l], wg_box + 0 * LAYSTR, b_box + 0 * CIN, pingA, H, Wd, lw,
        0, 0, nullptr, nullptr, nullptr);
    conv3x3_wmma_kernel<0, CIN, _Float16><<<g256, blk, 0, stream>>>(
        pingA, wg_box + 1 * LAYSTR, b_box + 1 * CIN, pingB, H, Wd, lw,
        0, 0, nullptr, nullptr, nullptr);
    conv3x3_wmma_kernel<0, CIN, _Float16><<<g256, blk, 0, stream>>>(
        pingB, wg_box + 2 * LAYSTR, b_box + 2 * CIN, pingA, H, Wd, lw,
        0, 0, nullptr, nullptr, nullptr);
    conv3x3_wmma_kernel<0, CIN, _Float16><<<g256, blk, 0, stream>>>(
        pingA, wg_box + 3 * LAYSTR, b_box + 3 * CIN, pingB, H, Wd, lw,
        0, 0, nullptr, nullptr, nullptr);
    conv3x3_wmma_kernel<2, 36, _Float16><<<grg, blk, 0, stream>>>(
        pingB, wg_rg, b_reg, regbuf, H, Wd, lw,
        l, ROFF[l], nullptr, nullptr, nullptr);
  }

  decode_nms_kernel<<<NIMG, 256, 0, stream>>>(cc, cinfo, cscore, regbuf, nmss,
                                              (float*)d_out);
  (void)in_sizes; (void)n_in; (void)out_size; (void)ws_size;
}
